// MPNN_LSTM_55259049230849
// MI455X (gfx1250) — compile-verified, matrix operand order checked by reading, not run on hardware
//
#include <hip/hip_runtime.h>

#define NN 100000
#define NE 1600000
#define HID 32
#define BN_EPS 1e-5f

typedef __attribute__((ext_vector_type(16))) _Float16 v16h;
typedef __attribute__((ext_vector_type(8)))  float    v8f;

__device__ __forceinline__ float sigmoidf_(float x) { return 1.0f / (1.0f + __expf(-x)); }

// ---------------- CDNA5 WMMA helpers (wave32 layouts, ISA 7.12.2) ----------------
// A (16x32 f16): lane L holds row M = L%16; half = L/16 selects K groups:
//   a[0..7]  = A[row][8h .. 8h+7],  a[8..15] = A[row][16+8h .. 16+8h+7]
__device__ __forceinline__ v16h a_frag_from_f32(const float* rowptr, int half) {
  const float* p0 = rowptr + 8 * half;
  const float* p1 = rowptr + 16 + 8 * half;
  v16h a;
#pragma unroll
  for (int i = 0; i < 8; ++i) {
    a[i]     = (_Float16)p0[i];
    a[8 + i] = (_Float16)p1[i];
  }
  return a;
}

// K index inside a 16x16x32 B tile for fragment element i (i = 2v+odd), given half.
__device__ __forceinline__ int b_k_of(int i, int half) {
  int v = i >> 1, odd = i & 1;
  return ((v < 4) ? (2 * v + 8 * half) : (16 + 2 * (v - 4) + 8 * half)) + odd;
}

__device__ __forceinline__ v8f wmma16(v16h a, v16h b, v8f c) {
  // emits v_wmma_f32_16x16x32_f16
  return __builtin_amdgcn_wmma_f32_16x16x32_f16(false, a, false, b, (short)0, c, false, false);
}

// ---------------- utility kernels ----------------
__global__ void k_fill(float* p, float v, int n) {
  int i = blockIdx.x * blockDim.x + threadIdx.x;
  if (i < n) p[i] = v;
}

__global__ void k_deg(const int* __restrict__ dst, const float* __restrict__ ew,
                      float* __restrict__ deg, int E) {
  int e = blockIdx.x * blockDim.x + threadIdx.x;
  if (e < E) atomicAdd(&deg[dst[e]], ew[e]);
}

__global__ void k_colsum(const float* __restrict__ x, float* __restrict__ xsum, int n) {
  int f = threadIdx.x & 31;
  int row = blockIdx.x * (blockDim.x >> 5) + (threadIdx.x >> 5);
  int stride = gridDim.x * (blockDim.x >> 5);
  float s = 0.f;
  for (int r = row; r < n; r += stride) s += x[r * HID + f];
  atomicAdd(&xsum[f], s);
}

// ---------------- GCN: XW = X(Nx32) @ W(32x32), WMMA f16 ----------------
// B fragments pre-swizzled per lane: sF[tile*32 + lane] = 16 contiguous f16 (b128 loads)
__global__ void k_gemm32(const float* __restrict__ X, const float* __restrict__ W,
                         float* __restrict__ XW, int n) {
  __shared__ v16h sF[2 * 32];
  _Float16* sFh = (_Float16*)sF;
  for (int idx = threadIdx.x; idx < 2 * 32 * 16; idx += blockDim.x) {
    int t = idx >> 9, lane = (idx >> 4) & 31, i = idx & 15;
    int col = lane & 15, half = lane >> 4;
    int k = b_k_of(i, half);
    sFh[idx] = (_Float16)W[k * 32 + t * 16 + col];  // B[k][n] = W[k][n]
  }
  __syncthreads();
  int lane = threadIdx.x & 31;
  int row0 = (blockIdx.x * (blockDim.x >> 5) + (threadIdx.x >> 5)) * 16;
  if (row0 >= n) return;
  int col = lane & 15, half = lane >> 4;
  v16h a = a_frag_from_f32(X + (row0 + col) * 32, half);
  v8f z = {};
  v8f acc0 = wmma16(a, sF[lane], z);
  v8f acc1 = wmma16(a, sF[32 + lane], z);
#pragma unroll
  for (int j = 0; j < 8; ++j) {
    int r = row0 + j + 8 * half;
    XW[r * 32 + col]      = acc0[j];
    XW[r * 32 + 16 + col] = acc1[j];
  }
}

// ---------------- edge scatter: agg[dst] += xw[src] * norm (L2-resident atomics) --
__global__ void k_scatter(const float* __restrict__ xw, const int* __restrict__ src,
                          const int* __restrict__ dst, const float* __restrict__ ew,
                          const float* __restrict__ deg, float* __restrict__ agg, int E) {
  int gid = blockIdx.x * blockDim.x + threadIdx.x;   // E*32 = 51.2M < 2^31
  int e = gid >> 5, f = gid & 31;
  if (e >= E) return;
  int s = src[e], d = dst[e];
  float norm = ew[e] * rsqrtf(deg[s]) * rsqrtf(deg[d]);
  atomicAdd(&agg[d * 32 + f], xw[s * 32 + f] * norm);
}

// ---------------- relu(agg + xw/deg + b) -> y, plus BN sum/sumsq ----------------
__global__ void k_bn_stats(const float* __restrict__ xw, const float* __restrict__ agg,
                           const float* __restrict__ deg, const float* __restrict__ bias,
                           float* __restrict__ y, float* __restrict__ sums, int n) {
  int f = threadIdx.x & 31;
  int row = blockIdx.x * (blockDim.x >> 5) + (threadIdx.x >> 5);
  int stride = gridDim.x * (blockDim.x >> 5);
  float b = bias[f], s = 0.f, ss = 0.f;
  for (int r = row; r < n; r += stride) {
    float v = agg[r * 32 + f] + xw[r * 32 + f] / deg[r] + b;
    v = fmaxf(v, 0.f);
    y[r * 32 + f] = v;
    s += v; ss += v * v;
  }
  atomicAdd(&sums[f], s);
  atomicAdd(&sums[32 + f], ss);
}

__global__ void k_bn_apply(float* __restrict__ y, const float* __restrict__ sums,
                           const float* __restrict__ g, const float* __restrict__ be, int n) {
  int i = blockIdx.x * blockDim.x + threadIdx.x;
  if (i >= n * 32) return;
  int f = i & 31;
  float inv = 1.0f / n;
  float mu = sums[f] * inv;
  float var = sums[32 + f] * inv - mu * mu;
  y[i] = (y[i] - mu) * rsqrtf(var + BN_EPS) * g[f] + be[f];
}

// gate column offset for compact tile index t in {0..5} -> {0,16,64,80,96,112}
__device__ __forceinline__ int gate_n0(int t) {
  return ((t < 2) ? 0 : (t < 4) ? 64 : 96) + (t & 1) * 16;
}

// shared LSTM epilogue: hn = sigmoid(o) * tanh(sigmoid(i) * tanh(g))
__device__ __forceinline__ void lstm_epilogue(const v8f* acc, const float* bih,
                                              const float* bhh, float* hn, int row0,
                                              int col, int half) {
#pragma unroll
  for (int t = 0; t < 2; ++t) {
    int h = t * 16 + col;
    float bi = bih[h] + bhh[h];
    float bg = bih[64 + h] + bhh[64 + h];
    float bo = bih[96 + h] + bhh[96 + h];
#pragma unroll
    for (int j = 0; j < 8; ++j) {
      float iv = sigmoidf_(acc[t][j] + bi);
      float gv = tanhf(acc[2 + t][j] + bg);
      float ov = sigmoidf_(acc[4 + t][j] + bo);
      hn[(row0 + j + 8 * half) * 32 + h] = ov * tanhf(iv * gv);
    }
  }
}

// ---------------- LSTM1: gates = [h1,h2](Nx64) @ wih1^T; f-gate dead (c0=0) ------
__global__ void k_lstm1(const float* __restrict__ h1, const float* __restrict__ h2,
                        const float* __restrict__ wih, const float* __restrict__ bih,
                        const float* __restrict__ bhh, float* __restrict__ hn, int n) {
  __shared__ v16h sF[12 * 32];   // frag = kstep*6 + tile
  _Float16* sFh = (_Float16*)sF;
  for (int idx = threadIdx.x; idx < 12 * 32 * 16; idx += blockDim.x) {
    int f = idx >> 9, lane = (idx >> 4) & 31, i = idx & 15;
    int s = f / 6, t = f % 6;
    int col = lane & 15, half = lane >> 4;
    int k = b_k_of(i, half) + 32 * s;
    int g = gate_n0(t) + col;
    sFh[idx] = (_Float16)wih[g * 64 + k];   // B[k][g] = wih^T
  }
  __syncthreads();
  int lane = threadIdx.x & 31;
  int row0 = (blockIdx.x * (blockDim.x >> 5) + (threadIdx.x >> 5)) * 16;
  if (row0 >= n) return;
  int col = lane & 15, half = lane >> 4;
  v8f zero = {};
  v8f acc[6];
#pragma unroll
  for (int t = 0; t < 6; ++t) acc[t] = zero;
  {
    v16h a = a_frag_from_f32(h1 + (row0 + col) * 32, half);   // K 0..31
#pragma unroll
    for (int t = 0; t < 6; ++t) acc[t] = wmma16(a, sF[t * 32 + lane], acc[t]);
  }
  {
    v16h a = a_frag_from_f32(h2 + (row0 + col) * 32, half);   // K 32..63
#pragma unroll
    for (int t = 0; t < 6; ++t) acc[t] = wmma16(a, sF[(6 + t) * 32 + lane], acc[t]);
  }
  lstm_epilogue(acc, bih, bhh, hn, row0, col, half);
}

// ---------------- LSTM2: gates = hn1(Nx32) @ wih2^T ----------------
__global__ void k_lstm2(const float* __restrict__ hin, const float* __restrict__ wih,
                        const float* __restrict__ bih, const float* __restrict__ bhh,
                        float* __restrict__ hn, int n) {
  __shared__ v16h sF[6 * 32];
  _Float16* sFh = (_Float16*)sF;
  for (int idx = threadIdx.x; idx < 6 * 32 * 16; idx += blockDim.x) {
    int t = idx >> 9, lane = (idx >> 4) & 31, i = idx & 15;
    int col = lane & 15, half = lane >> 4;
    int k = b_k_of(i, half);
    int g = gate_n0(t) + col;
    sFh[idx] = (_Float16)wih[g * 32 + k];   // B[k][g] = wih2^T
  }
  __syncthreads();
  int lane = threadIdx.x & 31;
  int row0 = (blockIdx.x * (blockDim.x >> 5) + (threadIdx.x >> 5)) * 16;
  if (row0 >= n) return;
  int col = lane & 15, half = lane >> 4;
  v8f zero = {};
  v8f acc[6];
#pragma unroll
  for (int t = 0; t < 6; ++t) acc[t] = zero;
  v16h a = a_frag_from_f32(hin + (row0 + col) * 32, half);
#pragma unroll
  for (int t = 0; t < 6; ++t) acc[t] = wmma16(a, sF[t * 32 + lane], acc[t]);
  lstm_epilogue(acc, bih, bhh, hn, row0, col, half);
}

// ---------------- fc1: z = relu([hn1,hn2,xmean](Nx96) @ fw1^T + fb1) ----------------
__global__ void k_fc1(const float* __restrict__ hn1, const float* __restrict__ hn2,
                      const float* __restrict__ xsum, const float* __restrict__ fw1,
                      const float* __restrict__ fb1, float* __restrict__ z, int n) {
  __shared__ v16h sF[6 * 32];   // frag = kstep*2 + tile
  __shared__ float sXM[32];
  _Float16* sFh = (_Float16*)sF;
  for (int idx = threadIdx.x; idx < 6 * 32 * 16; idx += blockDim.x) {
    int f = idx >> 9, lane = (idx >> 4) & 31, i = idx & 15;
    int s = f >> 1, t = f & 1;
    int col = lane & 15, half = lane >> 4;
    int k = b_k_of(i, half) + 32 * s;
    sFh[idx] = (_Float16)fw1[(t * 16 + col) * 96 + k];   // B[k][o] = fw1^T
  }
  if (threadIdx.x < 32) sXM[threadIdx.x] = xsum[threadIdx.x] * (1.0f / NN);
  __syncthreads();
  int lane = threadIdx.x & 31;
  int row0 = (blockIdx.x * (blockDim.x >> 5) + (threadIdx.x >> 5)) * 16;
  if (row0 >= n) return;
  int col = lane & 15, half = lane >> 4;
  v8f acc0 = {}, acc1 = {};
  {
    v16h a = a_frag_from_f32(hn1 + (row0 + col) * 32, half);      // K 0..31
    acc0 = wmma16(a, sF[lane], acc0);
    acc1 = wmma16(a, sF[32 + lane], acc1);
  }
  {
    v16h a = a_frag_from_f32(hn2 + (row0 + col) * 32, half);      // K 32..63
    acc0 = wmma16(a, sF[64 + lane], acc0);
    acc1 = wmma16(a, sF[96 + lane], acc1);
  }
  {
    v16h a = a_frag_from_f32(sXM, half);                          // K 64..95 (broadcast row)
    acc0 = wmma16(a, sF[128 + lane], acc0);
    acc1 = wmma16(a, sF[160 + lane], acc1);
  }
  float bb0 = fb1[col], bb1 = fb1[16 + col];
#pragma unroll
  for (int j = 0; j < 8; ++j) {
    int r = row0 + j + 8 * half;
    z[r * 32 + col]      = fmaxf(acc0[j] + bb0, 0.f);
    z[r * 32 + 16 + col] = fmaxf(acc1[j] + bb1, 0.f);
  }
}

// ---------------- fc2: out = z(Nx32) @ fw2^T + fb2 (matrix-vector, VALU) -----------
__global__ void k_fc2(const float* __restrict__ z, const float* __restrict__ fw2,
                      const float* __restrict__ fb2, float* __restrict__ out, int n) {
  int i = blockIdx.x * blockDim.x + threadIdx.x;
  if (i >= n) return;
  float s = fb2[0];
#pragma unroll
  for (int k = 0; k < 32; ++k) s += z[i * 32 + k] * fw2[k];
  out[i] = s;
}

extern "C" void kernel_launch(void* const* d_in, const int* in_sizes, int n_in,
                              void* d_out, int out_size, void* d_ws, size_t ws_size,
                              hipStream_t stream) {
  const int N = NN, E = NE;
  const float* x    = (const float*)d_in[0];
  const int*   ei   = (const int*)d_in[1];
  const float* ew   = (const float*)d_in[2];
  const float* w1   = (const float*)d_in[3];
  const float* b1   = (const float*)d_in[4];
  const float* g1   = (const float*)d_in[5];
  const float* be1  = (const float*)d_in[6];
  const float* w2   = (const float*)d_in[7];
  const float* b2   = (const float*)d_in[8];
  const float* g2   = (const float*)d_in[9];
  const float* be2  = (const float*)d_in[10];
  const float* wih1 = (const float*)d_in[11];
  // d_in[12] = whh1 unused: h0 == 0
  const float* bih1 = (const float*)d_in[13];
  const float* bhh1 = (const float*)d_in[14];
  const float* wih2 = (const float*)d_in[15];
  // d_in[16] = whh2 unused
  const float* bih2 = (const float*)d_in[17];
  const float* bhh2 = (const float*)d_in[18];
  const float* fw1  = (const float*)d_in[19];
  const float* fb1  = (const float*)d_in[20];
  const float* fw2  = (const float*)d_in[21];
  const float* fb2  = (const float*)d_in[22];
  const int* srcI = ei;
  const int* dstI = ei + E;

  float* ws   = (float*)d_ws;
  float* deg  = ws;                    // N
  float* xw   = deg + N;               // N*32
  float* agg  = xw  + (size_t)N * 32;  // N*32
  float* h1   = agg + (size_t)N * 32;  // N*32
  float* h2   = h1  + (size_t)N * 32;  // N*32
  float* hn1  = h2  + (size_t)N * 32;  // N*32
  float* hn2  = hn1 + (size_t)N * 32;  // N*32
  float* sums = hn2 + (size_t)N * 32;  // 64 (BN sum/sumsq)
  float* xsum = sums + 64;             // 32 (column sums of x)
  float* z    = xw;                    // reuse xw for fc1 output

  const int TB = 256;
  const int nodeElemBlocks = (N * 32 + TB - 1) / TB;
  const int tileBlocks = (N / 16 + 7) / 8;       // 16-row tiles, 8 waves/block
  const int edgeBlocks = (E + TB - 1) / TB;
  const int scatBlocks = (E * 32 + TB - 1) / TB; // feature-per-lane

  // init
  k_fill<<<(N + TB - 1) / TB, TB, 0, stream>>>(deg, 1.0f, N);        // self-loop weight
  k_fill<<<nodeElemBlocks, TB, 0, stream>>>(agg, 0.f, N * 32);
  k_fill<<<1, 96, 0, stream>>>(sums, 0.f, 96);                        // sums + xsum
  k_deg<<<edgeBlocks, TB, 0, stream>>>(dstI, ew, deg, E);
  k_colsum<<<512, TB, 0, stream>>>(x, xsum, N);

  // GCN layer 1
  k_gemm32<<<tileBlocks, TB, 0, stream>>>(x, w1, xw, N);
  k_scatter<<<scatBlocks, TB, 0, stream>>>(xw, srcI, dstI, ew, deg, agg, E);
  k_bn_stats<<<1024, TB, 0, stream>>>(xw, agg, deg, b1, h1, sums, N);
  k_bn_apply<<<nodeElemBlocks, TB, 0, stream>>>(h1, sums, g1, be1, N);

  // GCN layer 2
  k_fill<<<nodeElemBlocks, TB, 0, stream>>>(agg, 0.f, N * 32);
  k_fill<<<1, 64, 0, stream>>>(sums, 0.f, 64);
  k_gemm32<<<tileBlocks, TB, 0, stream>>>(h1, w2, xw, N);
  k_scatter<<<scatBlocks, TB, 0, stream>>>(xw, srcI, dstI, ew, deg, agg, E);
  k_bn_stats<<<1024, TB, 0, stream>>>(xw, agg, deg, b2, h2, sums, N);
  k_bn_apply<<<nodeElemBlocks, TB, 0, stream>>>(h2, sums, g2, be2, N);

  // LSTM stack + head
  k_lstm1<<<tileBlocks, TB, 0, stream>>>(h1, h2, wih1, bih1, bhh1, hn1, N);
  k_lstm2<<<tileBlocks, TB, 0, stream>>>(hn1, wih2, bih2, bhh2, hn2, N);
  k_fc1<<<tileBlocks, TB, 0, stream>>>(hn1, hn2, xsum, fw1, fb1, z, N);
  k_fc2<<<(N + TB - 1) / TB, TB, 0, stream>>>(z, fw2, fb2, (float*)d_out, N);
}
